// Net_81947976008362
// MI455X (gfx1250) — matrix operand order repaired and checked by reading, compiled, not run on hardware
//
#include <hip/hip_runtime.h>

// ---------------------------------------------------------------------------
// Types for CDNA5 WMMA bf16
// ---------------------------------------------------------------------------
typedef __attribute__((ext_vector_type(16))) __bf16 v16bf;
typedef __attribute__((ext_vector_type(8)))  __bf16 v8bf;
typedef __attribute__((ext_vector_type(8)))  float  v8f;

union ABFrag { v16bf v; v8bf h[2]; };

// ---------------------------------------------------------------------------
// Problem constants
// ---------------------------------------------------------------------------
#define BATCH       8192
#define IN_DIM      784
#define IN_PAD      832        // padded to multiple of 64
#define IMED        4096
#define CODE_DIM    8192
#define NSTRIPE     128
#define SDIM        64
#define K_NEUR      512
#define K_STRIPE    16

// GEMM tile config: 256 threads = 8 waves; each wave owns 32x64 of C.
#define BM 128
#define BN 128
#define BK 64
#define LDT (BK + 8)           // bf16 stride w/ pad (144 bytes, 16B aligned)

// ---------------------------------------------------------------------------
// gfx1250 async global->LDS copy (ASYNCcnt-tracked, no VGPR round trip).
// LDS operand is the low 32 bits of the generic pointer (ISA: LDS aperture
// maps addr[31:0] to the LDS offset).
// ---------------------------------------------------------------------------
__device__ __forceinline__ void async_copy_b128(const void* g, void* l) {
  asm volatile("global_load_async_to_lds_b128 %0, %1, off"
               :
               : "v"((unsigned)(unsigned long long)(uintptr_t)l),
                 "v"((unsigned long long)(uintptr_t)g)
               : "memory");
}
__device__ __forceinline__ void wait_async_zero() {
  asm volatile("s_wait_asynccnt 0x0" ::: "memory");
}

// ---------------------------------------------------------------------------
// fp32 -> bf16 conversion kernels
// ---------------------------------------------------------------------------
__global__ void cvt_f32_bf16(const float* __restrict__ src,
                             __bf16* __restrict__ dst, size_t n) {
  size_t i = (size_t)blockIdx.x * blockDim.x + threadIdx.x;
  if (i < n) dst[i] = (__bf16)src[i];
}

__global__ void cvt_pad_f32_bf16(const float* __restrict__ src,
                                 __bf16* __restrict__ dst,
                                 int rows, int cols, int ldd) {
  size_t i = (size_t)blockIdx.x * blockDim.x + threadIdx.x;
  size_t total = (size_t)rows * ldd;
  if (i >= total) return;
  int r = (int)(i / ldd);
  int c = (int)(i % ldd);
  float v = (c < cols) ? src[(size_t)r * cols + c] : 0.0f;
  dst[i] = (__bf16)v;
}

// ---------------------------------------------------------------------------
// bf16 NT GEMM with WMMA:  Y[M,N] = act(X[M,K] * W[N,K]^T + bias[N])
// Double-buffered LDS tiles filled by async global->LDS DMA; one barrier per
// K-slice; tile t+1 fill overlaps tile t WMMAs.
// ---------------------------------------------------------------------------
__global__ __launch_bounds__(256)
void gemm_bf16_nt(const __bf16* __restrict__ X, const __bf16* __restrict__ W,
                  const float* __restrict__ bias,
                  float* __restrict__ Yf, __bf16* __restrict__ Yb,
                  int M, int N, int K, int relu)
{
  __shared__ __bf16 As[2][BM * LDT];
  __shared__ __bf16 Bs[2][BN * LDT];

  const int tid  = threadIdx.x;
  const int lane = tid & 31;
  const int wave = tid >> 5;       // 0..7
  const int wm   = wave >> 1;      // 0..3 -> 32-row band
  const int wn   = wave & 1;       // 0..1 -> 64-col band

  const int bm = blockIdx.y * BM;
  const int bn = blockIdx.x * BN;

  v8f acc[2][4];
  #pragma unroll
  for (int i = 0; i < 2; ++i)
    #pragma unroll
    for (int j = 0; j < 4; ++j)
      #pragma unroll
      for (int r = 0; r < 8; ++r)
        acc[i][j][r] = 0.0f;

  // Tile staging: each thread DMAs 32 elems of A and 32 of B (4x b128 each).
  const int lrow = tid >> 1;           // 0..127
  const int lcol = (tid & 1) * 32;     // 0 or 32
  // Clamp OOB W rows (N=784 tail) so async reads stay in-bounds; the
  // epilogue's column guard discards their results.
  const int nrow = (bn + lrow < N) ? (bn + lrow) : (N - 1);

  const __bf16* gA = X + (size_t)(bm + lrow) * K + lcol;
  const __bf16* gB = W + (size_t)nrow * K + lcol;

  const int nt = K / BK;

  auto issue_tile = [&](int t, int p) {
    const __bf16* a = gA + (size_t)t * BK;
    const __bf16* b = gB + (size_t)t * BK;
    __bf16* la = &As[p][lrow * LDT + lcol];
    __bf16* lb = &Bs[p][lrow * LDT + lcol];
    #pragma unroll
    for (int c = 0; c < 4; ++c) {
      async_copy_b128(a + c * 8, la + c * 8);
      async_copy_b128(b + c * 8, lb + c * 8);
    }
  };

  issue_tile(0, 0);

  for (int t = 0; t < nt; ++t) {
    const int p = t & 1;
    wait_async_zero();        // my tile-t DMAs done
    __syncthreads();          // everyone's tile-t DMAs done; buf p^1 free
    if (t + 1 < nt) issue_tile(t + 1, p ^ 1);   // overlaps compute below

    #pragma unroll
    for (int kk = 0; kk < BK; kk += 32) {
      // A fragment (16x32): lanes 0-15 K={0..7,16..23}, lanes 16-31 K={8..15,24..31}
      ABFrag a[2];
      #pragma unroll
      for (int i = 0; i < 2; ++i) {
        const int r  = wm * 32 + i * 16 + (lane & 15);
        const int ko = kk + ((lane >> 4) ? 8 : 0);
        a[i].h[0] = *(const v8bf*)&As[p][r * LDT + ko];
        a[i].h[1] = *(const v8bf*)&As[p][r * LDT + ko + 16];
      }
      // B fragment (32x16): lane n holds col n; lanes 0-15 K={0..7,16..23}, lanes 16-31 K={8..15,24..31}
      ABFrag b[4];
      #pragma unroll
      for (int j = 0; j < 4; ++j) {
        const int c  = wn * 64 + j * 16 + (lane & 15);
        const int ko = kk + ((lane >> 4) ? 8 : 0);
        b[j].h[0] = *(const v8bf*)&Bs[p][c * LDT + ko];
        b[j].h[1] = *(const v8bf*)&Bs[p][c * LDT + ko + 16];
      }
      #pragma unroll
      for (int i = 0; i < 2; ++i)
        #pragma unroll
        for (int j = 0; j < 4; ++j)
          acc[i][j] = __builtin_amdgcn_wmma_f32_16x16x32_bf16(
              false, a[i].v, false, b[j].v, (short)0, acc[i][j], false, false);
    }
  }

  // Epilogue: C/D layout -> VGPR r holds M = r + 8*(lane>=16), N = lane&15.
  const int cn     = lane & 15;
  const int rshift = (lane >> 4) * 8;
  #pragma unroll
  for (int i = 0; i < 2; ++i) {
    #pragma unroll
    for (int j = 0; j < 4; ++j) {
      const int col = bn + wn * 64 + j * 16 + cn;
      if (col >= N) continue;
      const float bv = bias ? bias[col] : 0.0f;
      #pragma unroll
      for (int r = 0; r < 8; ++r) {
        const int row = bm + wm * 32 + i * 16 + rshift + r;
        float v = acc[i][j][r] + bv;
        if (relu) v = v > 0.0f ? v : 0.0f;
        if (Yf) Yf[(size_t)row * N + col] = v;
        if (Yb) Yb[(size_t)row * N + col] = (__bf16)v;
      }
    }
  }
}

// ---------------------------------------------------------------------------
// Per-row top-k select: neuron top-512 (of 8192) via bitwise binary search on
// uint-monotonic keys (values are relu'd, so >= 0), stripe top-16 (of 128) by
// exact rank.  Writes masked code as bf16 (input of GEMM3).
// One block (256 threads) per batch row.
// ---------------------------------------------------------------------------
__global__ __launch_bounds__(256)
void select_mask(const float* __restrict__ code,    // [B, CODE_DIM]
                 const float* __restrict__ scores,  // [B, NSTRIPE]
                 __bf16* __restrict__ out)          // [B, CODE_DIM]
{
  __shared__ unsigned      keys[CODE_DIM];
  __shared__ float         s_sc[NSTRIPE];
  __shared__ unsigned char s_act[NSTRIPE];
  __shared__ int           s_cnt;

  const int row = blockIdx.x;
  const int tid = threadIdx.x;
  const float* cr = code + (size_t)row * CODE_DIM;

  // ---- stripe routing mask: exact top-16 with index tie-break ----
  if (tid < NSTRIPE) s_sc[tid] = scores[(size_t)row * NSTRIPE + tid];
  __syncthreads();
  if (tid < NSTRIPE) {
    const float v = s_sc[tid];
    int rank = 0;
    #pragma unroll 4
    for (int j = 0; j < NSTRIPE; ++j) {
      const float o = s_sc[j];
      rank += (o > v) || (o == v && j < tid);
    }
    s_act[tid] = (rank < K_STRIPE) ? 1 : 0;
  }

  // ---- load row as uint keys (monotonic for non-negative floats) ----
  for (int i = tid; i < CODE_DIM; i += 256)
    keys[i] = __float_as_uint(cr[i]);
  __syncthreads();

  // ---- find the 512th-largest key: 32-step bitwise binary search ----
  unsigned prefix = 0u;
  for (int bit = 31; bit >= 0; --bit) {
    const unsigned cand = prefix | (1u << bit);
    if (tid == 0) s_cnt = 0;
    __syncthreads();
    int local = 0;
    for (int i = tid; i < CODE_DIM; i += 256)
      local += (keys[i] >= cand) ? 1 : 0;
    atomicAdd(&s_cnt, local);
    __syncthreads();
    if (s_cnt >= K_NEUR) prefix = cand;
    __syncthreads();
  }

  // ---- write masked bf16 code ----
  __bf16* orow = out + (size_t)row * CODE_DIM;
  for (int i = tid; i < CODE_DIM; i += 256) {
    const bool act = (keys[i] >= prefix) && s_act[i >> 6];
    const float v  = __uint_as_float(keys[i]);
    orow[i] = act ? (__bf16)v : (__bf16)0.0f;
  }
}

// ---------------------------------------------------------------------------
// Launch
// ---------------------------------------------------------------------------
static inline size_t alignup(size_t x) { return (x + 255) & ~(size_t)255; }

extern "C" void kernel_launch(void* const* d_in, const int* in_sizes, int n_in,
                              void* d_out, int out_size, void* d_ws, size_t ws_size,
                              hipStream_t stream) {
  (void)in_sizes; (void)n_in; (void)out_size; (void)ws_size;

  const float* x  = (const float*)d_in[0];
  const float* W1 = (const float*)d_in[1];
  const float* b1 = (const float*)d_in[2];
  const float* W2 = (const float*)d_in[3];
  const float* b2 = (const float*)d_in[4];
  const float* Wr = (const float*)d_in[5];
  const float* br = (const float*)d_in[6];
  const float* W3 = (const float*)d_in[7];
  const float* b3 = (const float*)d_in[8];
  const float* W4 = (const float*)d_in[9];
  const float* b4 = (const float*)d_in[10];

  char* ws = (char*)d_ws;
  size_t off = 0;
  __bf16* W1b = (__bf16*)(ws + off); off = alignup(off + (size_t)IMED * IN_PAD * 2);
  __bf16* W2b = (__bf16*)(ws + off); off = alignup(off + (size_t)CODE_DIM * IMED * 2);
  __bf16* Wrb = (__bf16*)(ws + off); off = alignup(off + (size_t)NSTRIPE * IMED * 2);
  __bf16* W3b = (__bf16*)(ws + off); off = alignup(off + (size_t)IMED * CODE_DIM * 2);
  __bf16* W4b = (__bf16*)(ws + off); off = alignup(off + (size_t)IN_DIM * IMED * 2);
  __bf16* Xb  = (__bf16*)(ws + off); off = alignup(off + (size_t)BATCH * IN_PAD * 2);
  __bf16* h1b = (__bf16*)(ws + off); off = alignup(off + (size_t)BATCH * IMED * 2);
  float*  sco = (float *)(ws + off); off = alignup(off + (size_t)BATCH * NSTRIPE * 4);
  float*  cod = (float *)(ws + off); off = alignup(off + (size_t)BATCH * CODE_DIM * 4);
  __bf16* codb= (__bf16*)(ws + off); off = alignup(off + (size_t)BATCH * CODE_DIM * 2);
  __bf16* db  = (__bf16*)(ws + off); off = alignup(off + (size_t)BATCH * IMED * 2);

  const int T = 256;
  // --- fp32 -> bf16 (with K padding for the 784-dim operands) ---
  {
    size_t n = (size_t)BATCH * IN_PAD;
    cvt_pad_f32_bf16<<<dim3((unsigned)((n + T - 1) / T)), T, 0, stream>>>(x, Xb, BATCH, IN_DIM, IN_PAD);
  }
  {
    size_t n = (size_t)IMED * IN_PAD;
    cvt_pad_f32_bf16<<<dim3((unsigned)((n + T - 1) / T)), T, 0, stream>>>(W1, W1b, IMED, IN_DIM, IN_PAD);
  }
  {
    size_t n = (size_t)CODE_DIM * IMED;
    cvt_f32_bf16<<<dim3((unsigned)((n + T - 1) / T)), T, 0, stream>>>(W2, W2b, n);
  }
  {
    size_t n = (size_t)NSTRIPE * IMED;
    cvt_f32_bf16<<<dim3((unsigned)((n + T - 1) / T)), T, 0, stream>>>(Wr, Wrb, n);
  }
  {
    size_t n = (size_t)IMED * CODE_DIM;
    cvt_f32_bf16<<<dim3((unsigned)((n + T - 1) / T)), T, 0, stream>>>(W3, W3b, n);
  }
  {
    size_t n = (size_t)IN_DIM * IMED;
    cvt_f32_bf16<<<dim3((unsigned)((n + T - 1) / T)), T, 0, stream>>>(W4, W4b, n);
  }

  const dim3 blk(256);
  const unsigned gy = BATCH / BM;   // 64

  // h1 = relu(x @ W1^T + b1)         -> bf16
  gemm_bf16_nt<<<dim3(IMED / BN, gy), blk, 0, stream>>>(
      Xb, W1b, b1, nullptr, h1b, BATCH, IMED, IN_PAD, 1);
  // scores = h1 @ Wr^T + br          -> f32
  gemm_bf16_nt<<<dim3(1, gy), blk, 0, stream>>>(
      h1b, Wrb, br, sco, nullptr, BATCH, NSTRIPE, IMED, 0);
  // code = relu(h1 @ W2^T + b2)      -> f32
  gemm_bf16_nt<<<dim3(CODE_DIM / BN, gy), blk, 0, stream>>>(
      h1b, W2b, b2, cod, nullptr, BATCH, CODE_DIM, IMED, 1);
  // top-k neuron + stripe masking    -> masked bf16 code
  select_mask<<<dim3(BATCH), blk, 0, stream>>>(cod, sco, codb);
  // d = relu(code_masked @ W3^T + b3) -> bf16
  gemm_bf16_nt<<<dim3(IMED / BN, gy), blk, 0, stream>>>(
      codb, W3b, b3, nullptr, db, BATCH, IMED, CODE_DIM, 1);
  // out = relu(d @ W4^T + b4)        -> f32 (direct to d_out, N=784 guarded)
  gemm_bf16_nt<<<dim3((IN_DIM + BN - 1) / BN, gy), blk, 0, stream>>>(
      db, W4b, b4, (float*)d_out, nullptr, BATCH, IN_DIM, IMED, 1);
}
